// GIN_8572754723378
// MI455X (gfx1250) — matrix operand order repaired and checked by reading, not compiled, not run on hardware
//
#include <hip/hip_runtime.h>

#define D 128
#define BN_EPS 1e-5f
#define ROWS_PER_BLOCK 32

typedef __bf16 bf16_t;
typedef bf16_t v16bf __attribute__((ext_vector_type(16)));
typedef float  v8f   __attribute__((ext_vector_type(8)));

__device__ __forceinline__ bf16_t f2bf(float f) { return (bf16_t)f; }

// ---------------------------------------------------------------- utilities
__global__ void zero_f(float* p, int n) {
    int i = blockIdx.x * blockDim.x + threadIdx.x;
    if (i < n) p[i] = 0.f;
}

__global__ void copy_f4(const float* __restrict__ in, float* __restrict__ out, int n4) {
    int i = blockIdx.x * blockDim.x + threadIdx.x;
    if (i < n4) ((float4*)out)[i] = ((const float4*)in)[i];
}

// ------------------------------------------------- edge scatter-add (GIN agg)
// One wave32 per edge; each lane moves a float4 (32 lanes * 4 = 128 features).
__global__ void scatter_add_edges(const float* __restrict__ h,
                                  const long long* __restrict__ src,
                                  const long long* __restrict__ dst,
                                  float* __restrict__ agg, int E) {
    int gid  = blockIdx.x * blockDim.x + threadIdx.x;
    int e    = gid >> 5;
    int lane = gid & 31;
    if (e >= E) return;
    long long s = src[e];
    long long d = dst[e];
    const float4 v = ((const float4*)(h + (size_t)s * D))[lane];
    float* p = agg + (size_t)d * D + lane * 4;
    atomicAdd(p + 0, v.x);
    atomicAdd(p + 1, v.y);
    atomicAdd(p + 2, v.z);
    atomicAdd(p + 3, v.w);
}

// --------------------------------- GEMM (bf16 WMMA) + bias + ReLU + BN stats
// out[n, o] = relu( sum_k A[n,k] * W[o,k] + bias[o] );  per-column sum/sumsq.
// A tile is staged f32 into LDS with global_load_async_to_lds_b128 (ASYNCcnt),
// W is staged bf16.  Each block: 32 rows x 128 cols; 8 waves = 2 row tiles x
// 4 column groups (2 col tiles each) -> 64 v_wmma per block.
__global__ __launch_bounds__(256)
void gemm_bias_relu_stats(const float* __restrict__ A,     // [N,128]  (h + agg)
                          const float* __restrict__ W,     // [128,128] row = out feature
                          const float* __restrict__ bias,  // [128]
                          float* __restrict__ out,         // [N,128]
                          float* __restrict__ colsum,      // [128]
                          float* __restrict__ colsumsq,    // [128]
                          int N) {
    __shared__ float  sAf[ROWS_PER_BLOCK][D + 8];  // f32 A tile (async-staged); 544B row stride
    __shared__ bf16_t sW[D][D + 8];                // bf16 W; 272B row stride (breaks bank period)

    const int tid     = threadIdx.x;
    const int rowBase = blockIdx.x * ROWS_PER_BLOCK;

    // ---- Stage W as bf16 in LDS (float4 vectorized; 32 float4 per row).
    for (int idx = tid; idx < (D * D) / 4; idx += 256) {
        int r = idx >> 5;
        int c = (idx & 31) * 4;
        float4 w4 = ((const float4*)W)[idx];
        union { unsigned long long u; bf16_t h[4]; } p;
        p.h[0] = f2bf(w4.x); p.h[1] = f2bf(w4.y); p.h[2] = f2bf(w4.z); p.h[3] = f2bf(w4.w);
        *(unsigned long long*)&sW[r][c] = p.u;
    }

    // ---- Stage A tile (f32) straight into LDS via the async engine.
    // saddr form: per-lane LDS byte address (VDST) + 32-bit voffset against SGPR base.
    const unsigned long long abase = (unsigned long long)A;
    for (int idx = tid; idx < (ROWS_PER_BLOCK * D) / 4; idx += 256) {
        int r = idx >> 5;                 // 32 16B-chunks per row
        int c = (idx & 31) * 4;
        int grow = rowBase + r;
        unsigned lds_off = (unsigned)(uintptr_t)&sAf[r][c];
        if (grow < N) {
            unsigned goff = ((unsigned)grow * (unsigned)D + (unsigned)c) * 4u;
            asm volatile("global_load_async_to_lds_b128 %0, %1, %2"
                         :: "v"(lds_off), "v"(goff), "s"(abase) : "memory");
        } else {
            *(float4*)&sAf[r][c] = make_float4(0.f, 0.f, 0.f, 0.f);
        }
    }
    // Hint the next block's A tile toward the caches (global_prefetch_b8).
    if (rowBase + ROWS_PER_BLOCK < N)
        __builtin_prefetch(A + (size_t)(rowBase + ROWS_PER_BLOCK) * D + (tid & 31) * 4, 0, 1);

    asm volatile("s_wait_asynccnt 0" ::: "memory");  // own wave's async loads done
    __syncthreads();                                  // all waves' loads visible

    const int wave = tid >> 5;
    const int lane = tid & 31;
    const int half = lane >> 4;      // lane group 0-15 vs 16-31
    const int m    = lane & 15;
    const int rowTile = wave & 1;                  // 2 row tiles of 16
    const int rloc    = rowTile * 16 + m;

    // A fragments (16x32 bf16, CDNA5 layout): lanes 0-15 hold K {0..7,16..23},
    // lanes 16-31 hold K {8..15,24..31} for row M = lane%16.  Convert f32->bf16
    // at fragment build; hoisted over col tiles.
    union Frag { v16bf v; uint4 q[2]; bf16_t h[16]; };
    Frag afrag[4];
#pragma unroll
    for (int kk = 0; kk < 4; ++kk) {
        int k0 = kk * 32;
        const float* p0 = &sAf[rloc][k0 + half * 8];
        const float* p1 = &sAf[rloc][k0 + 16 + half * 8];
#pragma unroll
        for (int j = 0; j < 8; ++j) {
            afrag[kk].h[j]     = f2bf(p0[j]);
            afrag[kk].h[8 + j] = f2bf(p1[j]);
        }
    }

    const bool full  = (rowBase + ROWS_PER_BLOCK) <= N;   // block-uniform fast path
    const int  cBase = (wave >> 1) * 2;                   // 4 col groups x 2 col tiles
#pragma unroll
    for (int ct = 0; ct < 2; ++ct) {
        const int c    = cBase + ct;
        const int ncol = c * 16 + m;               // this lane's output column (B/C/D layout)
        v8f acc = {0.f, 0.f, 0.f, 0.f, 0.f, 0.f, 0.f, 0.f};
#pragma unroll
        for (int kk = 0; kk < 4; ++kk) {
            int k0 = kk * 32;
            // B (32x16 bf16, K-major per column): lane holds column N = lane%16,
            // K {k0+half*8..+7, k0+16+half*8..+7} in sW[ncol][*] since B[k][n] = W[n][k].
            Frag bfrag;
            bfrag.q[0] = *(const uint4*)&sW[ncol][k0 + half * 8];
            bfrag.q[1] = *(const uint4*)&sW[ncol][k0 + 16 + half * 8];
            acc = __builtin_amdgcn_wmma_f32_16x16x32_bf16(
                false, afrag[kk].v, false, bfrag.v, (short)0, acc, false, false);
        }
        // Epilogue: bias + ReLU + store + per-column BN partial stats.
        // D layout: VGPR i -> row M = i + half*8, col N = lane%16.
        const float b  = bias[ncol];
        float s  = 0.f;
        float s2 = 0.f;
        const size_t rbase = (size_t)(rowBase + rowTile * 16 + half * 8);
        if (full) {
#pragma unroll
            for (int i = 0; i < 8; ++i) {
                float v = acc[i] + b;
                v = v > 0.f ? v : 0.f;
                out[(rbase + i) * D + ncol] = v;
                s  += v;
                s2 += v * v;
            }
        } else {
#pragma unroll
            for (int i = 0; i < 8; ++i) {
                float v = acc[i] + b;
                v = v > 0.f ? v : 0.f;
                if ((int)(rbase + i) < N) {
                    out[(rbase + i) * D + ncol] = v;
                    s  += v;
                    s2 += v * v;
                }
            }
        }
        atomicAdd(&colsum[ncol], s);
        atomicAdd(&colsumsq[ncol], s2);
    }
}

// ------------------------------------------------------------- BN finalize
__global__ void bn_finalize(const float* __restrict__ colsum,
                            const float* __restrict__ colsumsq,
                            float* __restrict__ mu, float* __restrict__ inv,
                            float invN) {
    int c = threadIdx.x;
    float m   = colsum[c] * invN;
    float var = colsumsq[c] * invN - m * m;
    mu[c]  = m;
    inv[c] = rsqrtf(var + BN_EPS);
}

// ------------------------------------------------------------- BN apply
__global__ void bn_apply(const float* __restrict__ h,
                         const float* __restrict__ mu, const float* __restrict__ inv,
                         const float* __restrict__ g,  const float* __restrict__ be,
                         float* __restrict__ out, int n4) {
    int i = blockIdx.x * blockDim.x + threadIdx.x;
    if (i >= n4) return;
    int c = (i & 31) * 4;                          // 32 float4 per 128-wide row
    float4 hv  = ((const float4*)h)[i];
    float4 muv = *(const float4*)(mu + c);
    float4 iv  = *(const float4*)(inv + c);
    float4 gv  = *(const float4*)(g + c);
    float4 bv  = *(const float4*)(be + c);
    float4 o;
    o.x = gv.x * (hv.x - muv.x) * iv.x + bv.x;
    o.y = gv.y * (hv.y - muv.y) * iv.y + bv.y;
    o.z = gv.z * (hv.z - muv.z) * iv.z + bv.z;
    o.w = gv.w * (hv.w - muv.w) * iv.w + bv.w;
    ((float4*)out)[i] = o;
}

// ---------------------------------------------------------------- launcher
extern "C" void kernel_launch(void* const* d_in, const int* in_sizes, int n_in,
                              void* d_out, int out_size, void* d_ws, size_t ws_size,
                              hipStream_t stream) {
    const float*     x   = (const float*)d_in[0];
    const long long* ei  = (const long long*)d_in[1];   // [2,E] int64
    const float*     W1  = (const float*)d_in[2];
    const float*     b1  = (const float*)d_in[3];
    const float*     g1  = (const float*)d_in[4];
    const float*     be1 = (const float*)d_in[5];
    const float*     W2  = (const float*)d_in[6];
    const float*     b2  = (const float*)d_in[7];
    const float*     g2  = (const float*)d_in[8];
    const float*     be2 = (const float*)d_in[9];

    const int N = in_sizes[0] / D;
    const int E = in_sizes[1] / 2;
    const long long* src = ei;
    const long long* dst = ei + E;

    const size_t nd = (size_t)N * D;
    float* ws    = (float*)d_ws;
    float* agg   = ws;             // [N,D]
    float* hmid  = ws + nd;        // [N,D]
    float* h1    = ws + 2 * nd;    // [N,D]
    float* stats = ws + 3 * nd;    // 8 x 128
    float* sum1 = stats,       *sumsq1 = stats + 128, *mu1 = stats + 256, *inv1 = stats + 384;
    float* sum2 = stats + 512, *sumsq2 = stats + 640, *mu2 = stats + 768, *inv2 = stats + 896;

    const int n4 = (int)(nd / 4);
    const dim3 blk(256);
    const dim3 gElem((n4 + 255) / 256);
    const dim3 gScat((E * 32 + 255) / 256);
    const dim3 gGemm((N + ROWS_PER_BLOCK - 1) / ROWS_PER_BLOCK);
    const float invN = 1.0f / (float)N;

    zero_f<<<dim3(4), blk, 0, stream>>>(stats, 1024);

    // ---- layer 1
    copy_f4<<<gElem, blk, 0, stream>>>(x, agg, n4);                 // agg = x (self term)
    scatter_add_edges<<<gScat, blk, 0, stream>>>(x, src, dst, agg, E);
    gemm_bias_relu_stats<<<gGemm, blk, 0, stream>>>(agg, W1, b1, hmid, sum1, sumsq1, N);
    bn_finalize<<<1, 128, 0, stream>>>(sum1, sumsq1, mu1, inv1, invN);
    bn_apply<<<gElem, blk, 0, stream>>>(hmid, mu1, inv1, g1, be1, h1, n4);

    // ---- layer 2
    copy_f4<<<gElem, blk, 0, stream>>>(h1, agg, n4);
    scatter_add_edges<<<gScat, blk, 0, stream>>>(h1, src, dst, agg, E);
    gemm_bias_relu_stats<<<gGemm, blk, 0, stream>>>(agg, W2, b2, hmid, sum2, sumsq2, N);
    bn_finalize<<<1, 128, 0, stream>>>(sum2, sumsq2, mu2, inv2, invN);
    bn_apply<<<gElem, blk, 0, stream>>>(hmid, mu2, inv2, g2, be2, (float*)d_out, n4);
}
